// NVnet_44641890075010
// MI455X (gfx1250) — matrix operand order repaired and checked by reading, not compiled, not run on hardware
//
#include <hip/hip_runtime.h>
#include <math.h>

typedef __attribute__((ext_vector_type(16))) __bf16 v16bf;
typedef __attribute__((ext_vector_type(8)))  float  v8f;

#define NNODES 8192
#define NGRAPH 128
#define NPG    64
#define CLDIM  512
#define ZDIM   1536
#define LAYERS 3
#define HEADS  8
#define HID    64
#define NEDGES 262144
#define NCLS   10
#define EPSV   1e-4f

__device__ __forceinline__ float lrelu(float x){ return x > 0.f ? x : 0.01f*x; }
__device__ __forceinline__ float sigm(float x){ return 1.f/(1.f+__expf(-x)); }

// ---------------------------------------------------------------------------
// Tiled bf16 WMMA GEMM: C[M,N] = A[M,K] @ B[K,N]  (row-major, fp32 in/out)
// block = 256 threads = 8 waves. Each wave owns a 16x64 output tile (4 WMMA
// tiles sharing one A operand); block covers a 128x64 strip. B tile staged
// in LDS *transposed* (Bt[col][k]) so each lane's 16-element B operand is a
// single contiguous 32-byte LDS read with k = hi*16 + j, the same k order as A.
// mode: 0 = none, 1 = +bias, 2 = +bias then leaky-relu.  N % 64 == 0.
// ---------------------------------------------------------------------------
__global__ __launch_bounds__(256)
void k_gemm_bf16(const float* __restrict__ A, const float* __restrict__ B,
                 const float* __restrict__ bias, float* __restrict__ C,
                 int M, int N, int K, int mode)
{
    __shared__ __attribute__((aligned(32))) __bf16 Bt[64*32];  // [c][k], 4KB
    const int tid  = threadIdx.x;
    const int wave = tid >> 5, lane = tid & 31;
    const int M0 = (blockIdx.y*8 + wave)*16;
    const int N0 = blockIdx.x*64;
    const int r  = M0 + (lane & 15);
    const int hi = lane >> 4;
    const int c  = lane & 15;
    v8f acc0 = {}, acc1 = {}, acc2 = {}, acc3 = {};
    for (int K0 = 0; K0 < K; K0 += 32) {
        // stage B[K0:K0+32, N0:N0+64] -> Bt[c][k]; coalesced float2 loads
        #pragma unroll
        for (int i = 0; i < 4; ++i) {
            int e  = tid + 256*i;          // 1024 float2 pairs
            int k  = e >> 5;               // 0..31
            int c2 = (e & 31) * 2;         // 0..62
            float2 bv = *(const float2*)&B[(size_t)(K0+k)*N + N0 + c2];
            Bt[(c2  )*32 + k] = (__bf16)bv.x;
            Bt[(c2+1)*32 + k] = (__bf16)bv.y;
        }
        __syncthreads();
        // A operand: lanes 0-15 rows M0..M0+15 K 0..15, lanes 16-31 K 16..31
        v16bf a;
        const float* Ar = A + (size_t)r*K + K0;
        #pragma unroll
        for (int v = 0; v < 8; ++v) {
            int ka = 2*v + hi*16;
            float2 a2 = *(const float2*)(Ar + ka);
            a[2*v]   = (__bf16)a2.x;
            a[2*v+1] = (__bf16)a2.y;
        }
        // B operands: contiguous 32B per lane, element j <-> K = hi*16 + j
        v16bf b0 = *(const v16bf*)&Bt[(c     )*32 + hi*16];
        v16bf b1 = *(const v16bf*)&Bt[(c + 16)*32 + hi*16];
        v16bf b2 = *(const v16bf*)&Bt[(c + 32)*32 + hi*16];
        v16bf b3 = *(const v16bf*)&Bt[(c + 48)*32 + hi*16];
        acc0 = __builtin_amdgcn_wmma_f32_16x16x32_bf16(false, a, false, b0, (short)0, acc0, false, false);
        acc1 = __builtin_amdgcn_wmma_f32_16x16x32_bf16(false, a, false, b1, (short)0, acc1, false, false);
        acc2 = __builtin_amdgcn_wmma_f32_16x16x32_bf16(false, a, false, b2, (short)0, acc2, false, false);
        acc3 = __builtin_amdgcn_wmma_f32_16x16x32_bf16(false, a, false, b3, (short)0, acc3, false, false);
        __syncthreads();
    }
    #pragma unroll
    for (int j = 0; j < 8; ++j) {
        int row = M0 + j + hi*8;
        float* Cr = C + (size_t)row*N + N0 + c;
        float v0 = acc0[j], v1 = acc1[j], v2 = acc2[j], v3 = acc3[j];
        if (mode >= 1) {
            v0 += bias[N0 + c];
            v1 += bias[N0 + c + 16];
            v2 += bias[N0 + c + 32];
            v3 += bias[N0 + c + 48];
        }
        if (mode == 2) { v0 = lrelu(v0); v1 = lrelu(v1); v2 = lrelu(v2); v3 = lrelu(v3); }
        Cr[0] = v0; Cr[16] = v1; Cr[32] = v2; Cr[48] = v3;
    }
}

// ---------------------------------------------------------------------------
// Dense per-(graph,head) attention aggregation as 64x64x64 WMMA matmuls.
// agg[i, h*64+d] = sum_j sigmoid(lrelu(s1[i,h]+s2[j,h])) * xh[j, h*64+d]
// One wave per 16x16 output tile; W materialized in registers.
// ---------------------------------------------------------------------------
__global__ __launch_bounds__(256)
void k_attn(const float* __restrict__ xh, const float* __restrict__ s1,
            const float* __restrict__ s2, float* __restrict__ agg)
{
    int wid  = blockIdx.x*8 + (threadIdx.x >> 5);   // 0..16383
    int lane = threadIdx.x & 31;
    int tile = wid & 15;  int gh = wid >> 4;
    int hh = gh & 7;      int g  = gh >> 3;
    int ti = tile >> 2,   tj = tile & 3;
    int base = g*NPG;
    int r = lane & 15, hi = lane >> 4, cc = lane & 15;
    int inode = base + ti*16 + r;
    float s1i = s1[inode*HEADS + hh];
    v8f acc = {};
    for (int kk = 0; kk < 64; kk += 32) {
        v16bf a, b;
        #pragma unroll
        for (int v = 0; v < 8; ++v) {
            int ka = 2*v + hi*16 + kk;
            float w0 = sigm(lrelu(s1i + s2[(base+ka  )*HEADS + hh]));
            float w1 = sigm(lrelu(s1i + s2[(base+ka+1)*HEADS + hh]));
            a[2*v]   = (__bf16)w0;
            a[2*v+1] = (__bf16)w1;
            int kb = 2*v + hi*16 + kk;
            b[2*v]   = (__bf16)xh[(size_t)(base+kb  )*CLDIM + hh*HID + tj*16 + cc];
            b[2*v+1] = (__bf16)xh[(size_t)(base+kb+1)*CLDIM + hh*HID + tj*16 + cc];
        }
        acc = __builtin_amdgcn_wmma_f32_16x16x32_bf16(false, a, false, b,
                                                      (short)0, acc, false, false);
    }
    #pragma unroll
    for (int j = 0; j < 8; ++j) {
        int node = base + ti*16 + j + hi*8;
        int col  = hh*HID + tj*16 + cc;
        agg[(size_t)node*CLDIM + col] = acc[j];
    }
}

// ---------------------------------------------------------------------------
// BatchNorm stats: coalesced row-chunk accumulation. Each block handles 256
// rows; each thread owns 2 columns in registers, then 4 global atomics.
// stats[0:512] = sum, stats[512:1024] = sum of squares (pre-zeroed).
// ---------------------------------------------------------------------------
__global__ __launch_bounds__(256)
void k_colstats(const float* __restrict__ h, float* __restrict__ stats)
{
    int tid = threadIdx.x;
    int r0  = blockIdx.x * 256;
    float s0 = 0.f, q0 = 0.f, s1v = 0.f, q1v = 0.f;
    for (int rr = 0; rr < 256; ++rr) {
        const float* row = h + (size_t)(r0 + rr)*CLDIM;
        float v0 = row[tid], v1 = row[tid + 256];
        s0 += v0; q0 += v0*v0;
        s1v += v1; q1v += v1*v1;
    }
    atomicAdd(&stats[tid],            s0);
    atomicAdd(&stats[tid + 256],      s1v);
    atomicAdd(&stats[512 + tid],      q0);
    atomicAdd(&stats[512 + tid + 256],q1v);
}

__global__ __launch_bounds__(256)
void k_bn(float* __restrict__ h, const float* __restrict__ stats,
          const float* __restrict__ g, const float* __restrict__ b)
{
    size_t i = (size_t)blockIdx.x*256 + threadIdx.x;
    int c = (int)(i & (CLDIM-1));
    float m  = stats[c] / (float)NNODES;
    float vv = stats[512 + c] / (float)NNODES - m*m;
    h[i] = (h[i] - m) * rsqrtf(vv + 1e-5f) * g[c] + b[c];
}

// ---------------------------------------------------------------------------
// GCN degree / scatter / finish
// ---------------------------------------------------------------------------
__global__ __launch_bounds__(256)
void k_deg(const int* __restrict__ ei, float* __restrict__ degp1)
{
    int e = blockIdx.x*256 + threadIdx.x;
    if (e < NEDGES) atomicAdd(&degp1[ei[NEDGES + e]], 1.f);
}

__global__ __launch_bounds__(256)
void k_dinv(float* __restrict__ degp1, float* __restrict__ dinv)
{
    int n = blockIdx.x*256 + threadIdx.x;
    if (n < NNODES) { float d = degp1[n] + 1.f; degp1[n] = d; dinv[n] = rsqrtf(d); }
}

__global__ __launch_bounds__(256)
void k_scatter(const float* __restrict__ h2, const int* __restrict__ ei,
               const float* __restrict__ dinv, float* __restrict__ agg)
{
    int e = blockIdx.x;
    int s = ei[e], d = ei[NEDGES + e];
    float cf = dinv[s] * dinv[d];
    const float* src = h2 + (size_t)s*CLDIM;
    float*       dst = agg + (size_t)d*CLDIM;
    for (int f = threadIdx.x; f < CLDIM; f += 256)
        atomicAdd(&dst[f], src[f]*cf);
}

__global__ __launch_bounds__(256)
void k_gcn_finish(const float* __restrict__ agg, const float* __restrict__ h2,
                  const float* __restrict__ degp1, const float* __restrict__ bias,
                  float* __restrict__ h, float* __restrict__ z, int layer)
{
    size_t i = (size_t)blockIdx.x*256 + threadIdx.x;
    int c = (int)(i & (CLDIM-1));
    int n = (int)(i >> 9);
    float v = agg[i] + h2[i]/degp1[n] + bias[c];
    v = fmaxf(v, 0.f);
    h[i] = v;
    z[(size_t)n*ZDIM + layer*CLDIM + c] = v;
}

// ---------------------------------------------------------------------------
// Link-prediction edge loss: wave per edge, float4 loads, shuffle reduction.
// ---------------------------------------------------------------------------
__global__ __launch_bounds__(256)
void k_edge(const float* __restrict__ z, const int* __restrict__ ei,
            float* __restrict__ tot, float* __restrict__ cnt, int neg)
{
    int e    = blockIdx.x*8 + (threadIdx.x >> 5);
    int lane = threadIdx.x & 31;
    int s = ei[e], d = ei[NEDGES + e];
    const float4* zs = (const float4*)(z + (size_t)s*ZDIM);
    const float4* zd = (const float4*)(z + (size_t)d*ZDIM);
    float acc = 0.f;
    #pragma unroll
    for (int i = 0; i < 12; ++i) {              // 1536 / (32 lanes * 4)
        float4 a = zs[lane + 32*i], b = zd[lane + 32*i];
        acc += a.x*b.x + a.y*b.y + a.z*b.z + a.w*b.w;
    }
    #pragma unroll
    for (int o = 16; o > 0; o >>= 1) acc += __shfl_xor(acc, o, 32);
    if (!lane) {
        float sg = sigm(acc);
        float l  = -__logf(EPSV + (neg ? (1.f - sg) : sg));
        int g = s >> 6;                          // batch[src] = src / 64
        atomicAdd(&tot[g], l);
        atomicAdd(&cnt[g], 1.f);
    }
}

// ---------------------------------------------------------------------------
// Attention scores s1/s2, final MLP head + softmax + per-graph pooling.
// ---------------------------------------------------------------------------
__global__ __launch_bounds__(256)
void k_s12(const float* __restrict__ xh, const float* __restrict__ phi,
           float* __restrict__ s1, float* __restrict__ s2)
{
    int t = blockIdx.x*256 + threadIdx.x;        // 65536 = 8192*8
    int n = t >> 3, hh = t & 7;
    const float* xr = xh + (size_t)n*CLDIM + hh*HID;
    const float* pa = phi + hh*2*HID;
    const float* pb = pa + HID;
    float a = 0.f, b = 0.f;
    for (int d = 0; d < HID; ++d) { float v = xr[d]; a += v*pa[d]; b += v*pb[d]; }
    s1[t] = a; s2[t] = b;
}

__global__ __launch_bounds__(256)
void k_fc2(const float* __restrict__ q, const float* __restrict__ W,
           const float* __restrict__ b, float* __restrict__ preds,
           float* __restrict__ ypacc)
{
    int n = blockIdx.x*256 + threadIdx.x;
    float lg[NCLS];
    #pragma unroll
    for (int c = 0; c < NCLS; ++c) lg[c] = b[c];
    const float* qr = q + (size_t)n*HID;
    for (int j = 0; j < HID; ++j) {
        float v = qr[j];
        #pragma unroll
        for (int c = 0; c < NCLS; ++c) lg[c] += v * W[j*NCLS + c];
    }
    float mx = lg[0];
    #pragma unroll
    for (int c = 1; c < NCLS; ++c) mx = fmaxf(mx, lg[c]);
    float sm = 0.f;
    #pragma unroll
    for (int c = 0; c < NCLS; ++c) { lg[c] = __expf(lg[c]-mx) + EPSV; sm += lg[c]; }
    int g = n >> 6;
    #pragma unroll
    for (int c = 0; c < NCLS; ++c) {
        float p = lg[c] / sm;
        preds[(size_t)n*NCLS + c] = p;
        atomicAdd(&ypacc[g*NCLS + c], p);
    }
}

__global__ __launch_bounds__(256)
void k_final(const float* __restrict__ ypacc, const float* __restrict__ accs,
             float* __restrict__ out)
{
    int tid = threadIdx.x;
    for (int i = tid; i < NGRAPH*NCLS; i += 256)
        out[i] = __logf(ypacc[i] / (float)NPG);
    if (!tid) {
        const float* totp = accs,       *cntp = accs + 128;
        const float* totn = accs + 256, *cntn = accs + 384;
        float L = 0.f;
        for (int g = 0; g < NGRAPH; ++g) {
            L += totp[g] / fmaxf(cntp[g], 1.f);
            L += totn[g] / fmaxf(cntn[g], 1.f);
        }
        out[NGRAPH*NCLS] = L / (float)NGRAPH;
    }
}

// ---------------------------------------------------------------------------
extern "C" void kernel_launch(void* const* d_in, const int* in_sizes, int n_in,
                              void* d_out, int out_size, void* d_ws, size_t ws_size,
                              hipStream_t stream)
{
    const float* x     = (const float*)d_in[0];
    const int*   ei    = (const int*)  d_in[1];
    const int*   nei   = (const int*)  d_in[2];
    // d_in[3] = batch, d_in[4] = midx: structure is implicit (node/64, dense pairs)
    const float* fcW   = (const float*)d_in[5];
    const float* fcb   = (const float*)d_in[6];
    const float* bng   = (const float*)d_in[7];
    const float* bnb   = (const float*)d_in[8];
    const float* convW = (const float*)d_in[9];
    const float* convb = (const float*)d_in[10];
    const float* featW = (const float*)d_in[11];
    const float* phi   = (const float*)d_in[12];
    const float* fc1W  = (const float*)d_in[13];
    const float* fc1b  = (const float*)d_in[14];
    const float* fc2W  = (const float*)d_in[15];
    const float* fc2b  = (const float*)d_in[16];
    float* out = (float*)d_out;

    float* ws    = (float*)d_ws;
    float* z     = ws;                                   // 8192*1536
    float* h     = z     + (size_t)NNODES*ZDIM;          // 8192*512
    float* t     = h     + (size_t)NNODES*CLDIM;         // 8192*512 (h2 / xh)
    float* u     = t     + (size_t)NNODES*CLDIM;         // 8192*512 (agg buffers)
    float* q     = u     + (size_t)NNODES*CLDIM;         // 8192*64
    float* s1    = q     + (size_t)NNODES*HID;           // 8192*8
    float* s2    = s1    + NNODES*HEADS;                 // 8192*8
    float* degp1 = s2    + NNODES*HEADS;                 // 8192
    float* dinv  = degp1 + NNODES;                       // 8192
    float* stats = dinv  + NNODES;                       // 1024: colsum, colsumsq
    float* accs  = stats + 1024;                         // 512: totp,cntp,totn,cntn
    float* ypacc = accs  + 512;                          // 1280

    dim3 blk(256);

    // h = x @ fcW + fcb                       (M=8192, N=512, K=128)
    k_gemm_bf16<<<dim3(CLDIM/64, NNODES/128), blk, 0, stream>>>(
        x, fcW, fcb, h, NNODES, CLDIM, 128, 1);

    // degrees (+1) and rsqrt
    hipMemsetAsync(degp1, 0, NNODES*sizeof(float), stream);
    k_deg <<<dim3(NEDGES/256), blk, 0, stream>>>(ei, degp1);
    k_dinv<<<dim3(NNODES/256), blk, 0, stream>>>(degp1, dinv);

    for (int l = 0; l < LAYERS; ++l) {
        hipMemsetAsync(stats, 0, 1024*sizeof(float), stream);
        k_colstats<<<dim3(NNODES/256), blk, 0, stream>>>(h, stats);
        k_bn<<<dim3(NNODES*CLDIM/256), blk, 0, stream>>>(
            h, stats, bng + l*CLDIM, bnb + l*CLDIM);
        k_gemm_bf16<<<dim3(CLDIM/64, NNODES/128), blk, 0, stream>>>(
            h, convW + (size_t)l*CLDIM*CLDIM, nullptr, t, NNODES, CLDIM, CLDIM, 0);
        hipMemsetAsync(u, 0, (size_t)NNODES*CLDIM*sizeof(float), stream);
        k_scatter<<<dim3(NEDGES), blk, 0, stream>>>(t, ei, dinv, u);
        k_gcn_finish<<<dim3(NNODES*CLDIM/256), blk, 0, stream>>>(
            u, t, degp1, convb + l*CLDIM, h, z, l);
    }

    // edge reconstruction loss (pos + neg)
    hipMemsetAsync(accs, 0, 512*sizeof(float), stream);
    k_edge<<<dim3(NEDGES/8), blk, 0, stream>>>(z, ei,  accs,       accs + 128, 0);
    k_edge<<<dim3(NEDGES/8), blk, 0, stream>>>(z, nei, accs + 256, accs + 384, 1);

    // xh = z @ featW                          (M=8192, N=512, K=1536)
    k_gemm_bf16<<<dim3(CLDIM/64, NNODES/128), blk, 0, stream>>>(
        z, featW, nullptr, t, NNODES, CLDIM, ZDIM, 0);

    k_s12<<<dim3(NNODES*HEADS/256), blk, 0, stream>>>(t, phi, s1, s2);

    // dense attention aggregation via WMMA   (128 graphs x 8 heads x 16 tiles)
    k_attn<<<dim3(NGRAPH*HEADS*16/8), blk, 0, stream>>>(t, s1, s2, u);

    // q = leaky_relu(u @ fc1W + fc1b)        (M=8192, N=64, K=512)
    k_gemm_bf16<<<dim3(HID/64, NNODES/128), blk, 0, stream>>>(
        u, fc1W, fc1b, q, NNODES, HID, CLDIM, 2);

    hipMemsetAsync(ypacc, 0, NGRAPH*NCLS*sizeof(float), stream);
    k_fc2<<<dim3(NNODES/256), blk, 0, stream>>>(q, fc2W, fc2b, out + NGRAPH*NCLS + 1, ypacc);
    k_final<<<dim3(1), blk, 0, stream>>>(ypacc, accs, out);
}
